// SoccerViT_14645838479546
// MI455X (gfx1250) — compile-verified, matrix operand order checked by reading, not run on hardware
//
#include <hip/hip_runtime.h>
#include <hip/hip_bf16.h>
#include <math.h>

typedef __attribute__((ext_vector_type(16))) _Float16 v16h;
typedef __attribute__((ext_vector_type(2)))  __fp16   pkh2;   // cvt_pkrtz result type
typedef __attribute__((ext_vector_type(8)))  float    v8f;

constexpr int kSEQ   = 1601;
constexpr int kHID   = 256;
constexpr int kDH    = 32;
constexpr int kNH    = 8;
constexpr int kBATCH = 4;
constexpr int kSEQT  = 101;          // ceil(1601/16)
constexpr int kSEQL  = 1616;         // kSEQT*16  (x / yln / hbuf row padding)
constexpr int kSEQA  = 1632;         // 51*32     (q / k rows, vT cols padding)
constexpr int kPATCH = 1600;
constexpr int kPDIM  = 300;
constexpr int kPDIMP = 320;          // padded K for patch projection

// ---------------------------------------------------------------------------
// Unpredicated 16x32 f32 tile -> f16 in CDNA5 16-bit A-matrix layout.
//   lane&15 selects row, lane>>4 selects K-half (kh = 0|8)
//   VGPR i<4: K = kh+2i,+1 ; VGPR i>=4: K = kh+16+2(i-4),+1
// Two contiguous 8-float runs per lane -> 4x b128 loads, 8x cvt_pk_rtz.
// Caller guarantees in-bounds (padded buffers).
// ---------------------------------------------------------------------------
__device__ __forceinline__ void pack2(v16h& r, int i, float a, float b) {
  pkh2 p = __builtin_amdgcn_cvt_pkrtz(a, b);
  r[2 * i]     = (_Float16)p.x;
  r[2 * i + 1] = (_Float16)p.y;
}

__device__ __forceinline__ v16h load_tile_f32(const float* __restrict__ A, long lda,
                                              int m0, int k0) {
  int lane = threadIdx.x & 31;
  const float* rowp = A + (long)(m0 + (lane & 15)) * lda + (k0 + ((lane >> 4) << 3));
  float4 q0 = *(const float4*)(rowp);
  float4 q1 = *(const float4*)(rowp + 4);
  float4 q2 = *(const float4*)(rowp + 16);
  float4 q3 = *(const float4*)(rowp + 20);
  v16h r;
  pack2(r, 0, q0.x, q0.y); pack2(r, 1, q0.z, q0.w);
  pack2(r, 2, q1.x, q1.y); pack2(r, 3, q1.z, q1.w);
  pack2(r, 4, q2.x, q2.y); pack2(r, 5, q2.z, q2.w);
  pack2(r, 6, q3.x, q3.y); pack2(r, 7, q3.z, q3.w);
  return r;
}

__device__ __forceinline__ v8f wmma_f16(v16h a, v16h b, v8f c) {
  return __builtin_amdgcn_wmma_f32_16x16x32_f16(false, a, false, b, (short)0, c,
                                                false, false);
}

// ---------------------------------------------------------------------------
// GEMM: out[.,n] = act(A[M,K] * W[N,K]^T + bias) (+res). All dims padded;
// no masking anywhere. One wave per 16x16 tile. ACT: 0 none, 1 gelu, 2 relu.
// ---------------------------------------------------------------------------
template <int K, int ACT, bool RES>
__global__ void gemm_kernel(float* __restrict__ out, long ldo,
                            const float* __restrict__ A, long lda,
                            const float* __restrict__ W, long ldw,
                            const float* __restrict__ bias,
                            const float* __restrict__ res, long ldr) {
  int m0 = blockIdx.x * 16;
  int n0 = blockIdx.y * 16;
  v8f acc = {};
#pragma unroll
  for (int k0 = 0; k0 < K; k0 += 32) {
    v16h a = load_tile_f32(A, lda, m0, k0);
    v16h b = load_tile_f32(W, ldw, n0, k0);
    acc = wmma_f16(a, b, acc);
  }
  int lane = threadIdx.x & 31;
  int n = n0 + (lane & 15);
  long mb = m0 + ((lane >> 4) << 3);
  float bv = bias[n];
#pragma unroll
  for (int r = 0; r < 8; ++r) {
    float v = acc[r] + bv;
    if (ACT == 1)      v = 0.5f * v * (1.0f + erff(v * 0.70710678118654752f));
    else if (ACT == 2) v = fmaxf(v, 0.0f);
    if (RES) v += res[(mb + r) * ldr + n];
    out[(mb + r) * ldo + n] = v;
  }
}

// ---------------------------------------------------------------------------
// Patch extraction + LN over 300 dims; writes zero-padded rows of 320.
// ---------------------------------------------------------------------------
__global__ void patch_ln_kernel(float* __restrict__ patches,
                                const float* __restrict__ images,
                                const float* __restrict__ g,
                                const float* __restrict__ b) {
  int row = blockIdx.x;                 // b*1600 + p
  int bt = row / kPATCH, p = row % kPATCH;
  int py = p / 40, px = p % 40;
  int lane = threadIdx.x;
  float vals[10];
  float s = 0.0f;
#pragma unroll
  for (int i = 0; i < 10; ++i) {
    int e = lane + 32 * i;
    float v = 0.0f;
    if (e < kPDIM) {
      int c = e % 3, t = e / 3, ix = t % 10, iy = t / 10;
      v = images[(((long)bt * 3 + c) * 400 + (py * 10 + iy)) * 400 + (px * 10 + ix)];
    }
    vals[i] = v; s += v;
  }
  for (int o = 16; o > 0; o >>= 1) s += __shfl_xor(s, o, 32);
  float mean = s * (1.0f / 300.0f);
  float vs = 0.0f;
#pragma unroll
  for (int i = 0; i < 10; ++i) {
    int e = lane + 32 * i;
    if (e < kPDIM) { float d = vals[i] - mean; vs += d * d; }
  }
  for (int o = 16; o > 0; o >>= 1) vs += __shfl_xor(vs, o, 32);
  float inv = rsqrtf(vs * (1.0f / 300.0f) + 1e-5f);
  float* outp = patches + (long)row * kPDIMP;
#pragma unroll
  for (int i = 0; i < 10; ++i) {
    int e = lane + 32 * i;
    if (e < kPDIM) outp[e] = (vals[i] - mean) * inv * g[e] + b[e];
  }
  if (lane < kPDIMP - kPDIM) outp[kPDIM + lane] = 0.0f;   // zero K padding
}

// zero-padded copy of patch_w: [256][300] -> [256][320]
__global__ void pad_w_kernel(float* __restrict__ wp, const float* __restrict__ w) {
  int n = blockIdx.x;
  int lane = threadIdx.x;
  for (int i = lane; i < kPDIMP; i += 32)
    wp[(long)n * kPDIMP + i] = (i < kPDIM) ? w[(long)n * kPDIM + i] : 0.0f;
}

// ---------------------------------------------------------------------------
// LayerNorm(256), one wave per row, float4 IO.
// ---------------------------------------------------------------------------
__global__ void ln256_kernel(float* __restrict__ out, const float* __restrict__ in,
                             const float* __restrict__ g, const float* __restrict__ b) {
  long row = blockIdx.x;
  int lane = threadIdx.x;
  const float4* x4 = (const float4*)(in + row * kHID) + lane * 2;
  float4 a = x4[0], c = x4[1];
  float s = a.x + a.y + a.z + a.w + c.x + c.y + c.z + c.w;
  for (int o = 16; o > 0; o >>= 1) s += __shfl_xor(s, o, 32);
  float mean = s * (1.0f / 256.0f);
  float vs = 0.0f;
  {
    float d;
    d = a.x - mean; vs += d * d;  d = a.y - mean; vs += d * d;
    d = a.z - mean; vs += d * d;  d = a.w - mean; vs += d * d;
    d = c.x - mean; vs += d * d;  d = c.y - mean; vs += d * d;
    d = c.z - mean; vs += d * d;  d = c.w - mean; vs += d * d;
  }
  for (int o = 16; o > 0; o >>= 1) vs += __shfl_xor(vs, o, 32);
  float inv = rsqrtf(vs * (1.0f / 256.0f) + 1e-5f);
  int c0 = lane * 8;
  const float4* g4 = (const float4*)(g + c0);
  const float4* b4 = (const float4*)(b + c0);
  float4 g0 = g4[0], g1 = g4[1], b0 = b4[0], b1 = b4[1];
  float4 o0, o1;
  o0.x = (a.x - mean) * inv * g0.x + b0.x;  o0.y = (a.y - mean) * inv * g0.y + b0.y;
  o0.z = (a.z - mean) * inv * g0.z + b0.z;  o0.w = (a.w - mean) * inv * g0.w + b0.w;
  o1.x = (c.x - mean) * inv * g1.x + b1.x;  o1.y = (c.y - mean) * inv * g1.y + b1.y;
  o1.z = (c.z - mean) * inv * g1.z + b1.z;  o1.w = (c.w - mean) * inv * g1.w + b1.w;
  float4* y4 = (float4*)(out + row * kHID) + lane * 2;
  y4[0] = o0; y4[1] = o1;
}

// ---------------------------------------------------------------------------
// Assemble: x[b,0]=cls+pos[0]; x[b,1..1600]=LN2(embed)+pos[s]; x[b,>=1601]=0
// grid (kSEQL, BATCH)
// ---------------------------------------------------------------------------
__global__ void assemble_kernel(float* __restrict__ x, const float* __restrict__ embed,
                                const float* __restrict__ cls,
                                const float* __restrict__ g, const float* __restrict__ b,
                                const float* __restrict__ pos) {
  int s = blockIdx.x, bt = blockIdx.y;
  int lane = threadIdx.x;
  float* xo = x + ((long)bt * kSEQL + s) * kHID;
  if (s >= kSEQ) {
#pragma unroll
    for (int i = 0; i < 8; ++i) xo[lane + 32 * i] = 0.0f;
    return;
  }
  const float* pp = pos + (long)s * kHID;
  if (s == 0) {
#pragma unroll
    for (int i = 0; i < 8; ++i) { int c = lane + 32 * i; xo[c] = cls[c] + pp[c]; }
    return;
  }
  const float* e = embed + ((long)bt * kPATCH + (s - 1)) * kHID;
  float vals[8];
  float sm = 0.0f;
#pragma unroll
  for (int i = 0; i < 8; ++i) { vals[i] = e[lane + 32 * i]; sm += vals[i]; }
  for (int o = 16; o > 0; o >>= 1) sm += __shfl_xor(sm, o, 32);
  float mean = sm * (1.0f / 256.0f);
  float vs = 0.0f;
#pragma unroll
  for (int i = 0; i < 8; ++i) { float d = vals[i] - mean; vs += d * d; }
  for (int o = 16; o > 0; o >>= 1) vs += __shfl_xor(vs, o, 32);
  float inv = rsqrtf(vs * (1.0f / 256.0f) + 1e-5f);
#pragma unroll
  for (int i = 0; i < 8; ++i) {
    int c = lane + 32 * i;
    xo[c] = (vals[i] - mean) * inv * g[c] + b[c] + pp[c];
  }
}

// ---------------------------------------------------------------------------
// QKV: one wave per (qtile, head, batch); 6 WMMAs. q,k: [(b*H+h)*kSEQA+s][32];
// v stored transposed: vT[(b*H+h)*32 + d][kSEQA] for contiguous PV B-loads.
// ---------------------------------------------------------------------------
__global__ void qkv_kernel(float* __restrict__ q, float* __restrict__ k,
                           float* __restrict__ vT, const float* __restrict__ yln,
                           const float* __restrict__ wq, const float* __restrict__ bq,
                           const float* __restrict__ wk, const float* __restrict__ bk,
                           const float* __restrict__ wv, const float* __restrict__ bv) {
  int qt = blockIdx.x, h = blockIdx.y, bt = blockIdx.z;
  int m0 = qt * 16;
  int lane = threadIdx.x & 31;
  const float* A = yln + (long)bt * kSEQL * kHID + h * kDH;        // lda = kHID
  v16h a = load_tile_f32(A, kHID, m0, 0);
  int n16 = lane & 15;
  long mb = m0 + ((lane >> 4) << 3);
  long qkbase = (long)(bt * kNH + h) * kSEQA * kDH;
  long vtbase = (long)(bt * kNH + h) * kDH * kSEQA;

#pragma unroll
  for (int nt = 0; nt < 2; ++nt) {
    int n = nt * 16 + n16;
    // Q
    {
      v16h bm = load_tile_f32(wq + (long)h * kDH * kDH, kDH, nt * 16, 0);
      v8f acc = {};
      acc = wmma_f16(a, bm, acc);
      float bias = bq[h * kDH + n];
      float* o = q + qkbase;
#pragma unroll
      for (int r = 0; r < 8; ++r) o[(mb + r) * kDH + n] = acc[r] + bias;
    }
    // K
    {
      v16h bm = load_tile_f32(wk + (long)h * kDH * kDH, kDH, nt * 16, 0);
      v8f acc = {};
      acc = wmma_f16(a, bm, acc);
      float bias = bk[h * kDH + n];
      float* o = k + qkbase;
#pragma unroll
      for (int r = 0; r < 8; ++r) o[(mb + r) * kDH + n] = acc[r] + bias;
    }
    // V (transposed store, contiguous in key dim -> 2x b128)
    {
      v16h bm = load_tile_f32(wv + (long)h * kDH * kDH, kDH, nt * 16, 0);
      v8f acc = {};
      acc = wmma_f16(a, bm, acc);
      float bias = bv[h * kDH + n];
      float* o = vT + vtbase + (long)n * kSEQA + mb;
      float4 lo = make_float4(acc[0] + bias, acc[1] + bias, acc[2] + bias, acc[3] + bias);
      float4 hi = make_float4(acc[4] + bias, acc[5] + bias, acc[6] + bias, acc[7] + bias);
      *(float4*)(o)     = lo;
      *(float4*)(o + 4) = hi;
    }
  }
}

// ---------------------------------------------------------------------------
// Fused attention: one wave per (qtile, head, batch).
// Pass1: 102 tiles of S=scale*QK^T -> LDS [16][1632] (cols>=1601 = -1e30)
// softmax: 2 lanes per row; Pass2: 51x2 WMMAs of P@V^T, += into x.
// Dynamic LDS = 16*1632*4 = 104448 B (<= 320KB/WG on CDNA5).
// ---------------------------------------------------------------------------
__global__ void attn_kernel(float* __restrict__ x, const float* __restrict__ q,
                            const float* __restrict__ kk,
                            const float* __restrict__ vT) {
  extern __shared__ float sc[];        // [16][kSEQA]
  int qt = blockIdx.x, h = blockIdx.y, bt = blockIdx.z;
  int lane = threadIdx.x & 31;
  long qkbase = (long)(bt * kNH + h) * kSEQA * kDH;
  const float* Q = q + qkbase;
  const float* K = kk + qkbase;
  const float* V = vT + (long)(bt * kNH + h) * kDH * kSEQA;
  int m0 = qt * 16;
  const float scale = 0.17677669529663687f;   // 1/sqrt(32)

  v16h aq = load_tile_f32(Q, kDH, m0, 0);

  int n16 = lane & 15;
  int mb = (lane >> 4) << 3;
  for (int kt = 0; kt < kSEQA / 16; ++kt) {
    v16h bkt = load_tile_f32(K, kDH, kt * 16, 0);
    v8f s = {};
    s = wmma_f16(aq, bkt, s);
    int n = kt * 16 + n16;
    bool valid = (n < kSEQ);
#pragma unroll
    for (int r = 0; r < 8; ++r)
      sc[(mb + r) * kSEQA + n] = valid ? s[r] * scale : -1e30f;
  }
  __syncthreads();

  {  // softmax: 2 lanes per row, 816 cols each
    int rrow = lane & 15, half = lane >> 4;
    float* rowp = sc + (long)rrow * kSEQA + half * (kSEQA / 2);
    float mx = -1e30f;
    for (int c = 0; c < kSEQA / 2; ++c) mx = fmaxf(mx, rowp[c]);
    mx = fmaxf(mx, __shfl_xor(mx, 16, 32));
    float sum = 0.0f;
    for (int c = 0; c < kSEQA / 2; ++c) {
      float e = __expf(rowp[c] - mx);
      rowp[c] = e; sum += e;
    }
    sum += __shfl_xor(sum, 16, 32);
    float inv = 1.0f / sum;
    for (int c = 0; c < kSEQA / 2; ++c) rowp[c] *= inv;
  }
  __syncthreads();

  v8f o0 = {}, o1 = {};
  for (int kt = 0; kt < kSEQA / 32; ++kt) {
    v16h ap = load_tile_f32(sc, kSEQA, 0, kt * 32);       // P tile from LDS
    v16h b0 = load_tile_f32(V, kSEQA, 0, kt * 32);        // V^T rows = dh 0..15
    o0 = wmma_f16(ap, b0, o0);
    v16h b1 = load_tile_f32(V, kSEQA, 16, kt * 32);       // dh 16..31
    o1 = wmma_f16(ap, b1, o1);
  }

#pragma unroll
  for (int r = 0; r < 8; ++r) {
    long xi = ((long)bt * kSEQL + (m0 + mb + r)) * kHID + h * kDH;
    x[xi + n16]      += o0[r];
    x[xi + 16 + n16] += o1[r];
  }
}

// Final tiny head: out[b,o] = sigmoid(h2[b] . w3[o] + b3[o]),  B=4, OUT=4
__global__ void head_final_kernel(float* __restrict__ out, const float* __restrict__ h2,
                                  const float* __restrict__ w3,
                                  const float* __restrict__ b3) {
  int lane = threadIdx.x;
  if (lane < 16) {
    int bt = lane >> 2, o = lane & 3;
    float s = b3[o];
    for (int c = 0; c < kHID; ++c) s += h2[bt * kHID + c] * w3[o * kHID + c];
    out[bt * 4 + o] = 1.0f / (1.0f + __expf(-s));
  }
}

// ---------------------------------------------------------------------------
struct BlockP {
  const float *bk, *bq, *bv, *ln1_b, *ln1_g, *ln2_b, *ln2_g;
  const float *mlp_b1, *mlp_b2, *mlp_w1, *mlp_w2, *wk, *wq, *wv;
};

extern "C" void kernel_launch(void* const* d_in, const int* in_sizes, int n_in,
                              void* d_out, int out_size, void* d_ws, size_t ws_size,
                              hipStream_t stream) {
  (void)in_sizes; (void)n_in; (void)out_size; (void)ws_size;
  // jax pytree (alphabetical dict key) flattening order.
  const float* images = (const float*)d_in[0];
  BlockP blk[6];
  int idx = 1;
  for (int i = 0; i < 6; ++i) {
    blk[i].bk     = (const float*)d_in[idx + 0];
    blk[i].bq     = (const float*)d_in[idx + 1];
    blk[i].bv     = (const float*)d_in[idx + 2];
    blk[i].ln1_b  = (const float*)d_in[idx + 3];
    blk[i].ln1_g  = (const float*)d_in[idx + 4];
    blk[i].ln2_b  = (const float*)d_in[idx + 5];
    blk[i].ln2_g  = (const float*)d_in[idx + 6];
    blk[i].mlp_b1 = (const float*)d_in[idx + 7];
    blk[i].mlp_b2 = (const float*)d_in[idx + 8];
    blk[i].mlp_w1 = (const float*)d_in[idx + 9];
    blk[i].mlp_w2 = (const float*)d_in[idx + 10];
    blk[i].wk     = (const float*)d_in[idx + 11];
    blk[i].wq     = (const float*)d_in[idx + 12];
    blk[i].wv     = (const float*)d_in[idx + 13];
    idx += 14;
  }
  const float* cls         = (const float*)d_in[idx++];
  const float* head_b1     = (const float*)d_in[idx++];
  const float* head_b2     = (const float*)d_in[idx++];
  const float* head_b3     = (const float*)d_in[idx++];
  const float* head_w1     = (const float*)d_in[idx++];
  const float* head_w2     = (const float*)d_in[idx++];
  const float* head_w3     = (const float*)d_in[idx++];
  const float* patch_b     = (const float*)d_in[idx++];
  const float* patch_ln1_b = (const float*)d_in[idx++];
  const float* patch_ln1_g = (const float*)d_in[idx++];
  const float* patch_ln2_b = (const float*)d_in[idx++];
  const float* patch_ln2_g = (const float*)d_in[idx++];
  const float* patch_w     = (const float*)d_in[idx++];
  const float* pos         = (const float*)d_in[idx++];

  // --- workspace layout (floats), everything padded so loads never mask ---
  float* ws = (float*)d_ws;
  const long ROWSL = (long)kBATCH * kSEQL;            // 6464 (mult of 16)
  float* x    = ws;                                   // 1,654,784
  float* yln  = x    + ROWSL * kHID;                  // 1,654,784
  float* qb   = yln  + ROWSL * kHID;                  // 1,671,168
  float* kb   = qb   + (long)kBATCH * kNH * kSEQA * kDH;
  float* vTb  = kb   + (long)kBATCH * kNH * kSEQA * kDH;
  float* hbuf = vTb  + (long)kBATCH * kNH * kSEQA * kDH;   // 6,619,136
  float* wpad = hbuf + ROWSL * 4 * kHID;              // 81,920
  float* h1   = wpad + (long)kHID * kPDIMP;           // 16 rows padded
  float* h2   = h1   + 16 * kHID;
  float* patches = hbuf;                              // alias (pre-blocks only)
  float* embed   = patches + (long)kBATCH * kPATCH * kPDIMP;

  const int attn_lds = 16 * kSEQA * 4;                // 104448 B

  // 0) pad patch_w into [256][320]
  pad_w_kernel<<<kHID, 32, 0, stream>>>(wpad, patch_w);
  // 1) patch extraction + LN1 (zero-padded K)
  patch_ln_kernel<<<kBATCH * kPATCH, 32, 0, stream>>>(patches, images,
                                                      patch_ln1_g, patch_ln1_b);
  // 2) embed = patches @ patch_w^T + patch_b   (M=6400, N=256, K=320)
  gemm_kernel<kPDIMP, 0, false><<<dim3(kBATCH * kPATCH / 16, kHID / 16), 32, 0,
                                  stream>>>(
      embed, kHID, patches, kPDIMP, wpad, kPDIMP, patch_b, nullptr, 0);
  // 3) LN2 + cls + pos -> x (padded rows zeroed)
  assemble_kernel<<<dim3(kSEQL, kBATCH), 32, 0, stream>>>(
      x, embed, cls, patch_ln2_g, patch_ln2_b, pos);
  // 4) transformer blocks
  for (int i = 0; i < 6; ++i) {
    ln256_kernel<<<(int)ROWSL, 32, 0, stream>>>(yln, x, blk[i].ln1_g, blk[i].ln1_b);
    qkv_kernel<<<dim3(kSEQT, kNH, kBATCH), 32, 0, stream>>>(
        qb, kb, vTb, yln, blk[i].wq, blk[i].bq, blk[i].wk, blk[i].bk,
        blk[i].wv, blk[i].bv);
    attn_kernel<<<dim3(kSEQT, kNH, kBATCH), 32, attn_lds, stream>>>(x, qb, kb, vTb);
    ln256_kernel<<<(int)ROWSL, 32, 0, stream>>>(yln, x, blk[i].ln2_g, blk[i].ln2_b);
    gemm_kernel<kHID, 1, false><<<dim3((int)ROWSL / 16, (4 * kHID) / 16), 32, 0,
                                  stream>>>(
        hbuf, 4 * kHID, yln, kHID, blk[i].mlp_w1, kHID, blk[i].mlp_b1, nullptr, 0);
    gemm_kernel<4 * kHID, 0, true><<<dim3((int)ROWSL / 16, kHID / 16), 32, 0,
                                     stream>>>(
        x, kHID, hbuf, 4 * kHID, blk[i].mlp_w2, 4 * kHID, blk[i].mlp_b2, x, kHID);
  }
  // 5) head on cls rows (stride kSEQL*kHID); rows 4..15 are padded garbage
  gemm_kernel<kHID, 2, false><<<dim3(1, kHID / 16), 32, 0, stream>>>(
      h1, kHID, x, (long)kSEQL * kHID, head_w1, kHID, head_b1, nullptr, 0);
  gemm_kernel<kHID, 2, false><<<dim3(1, kHID / 16), 32, 0, stream>>>(
      h2, kHID, h1, kHID, head_w2, kHID, head_b2, nullptr, 0);
  head_final_kernel<<<1, 32, 0, stream>>>((float*)d_out, h2, head_w3, head_b3);
}
